// ShiftedProxActLayer_39273180955314
// MI455X (gfx1250) — matrix operand order repaired and checked by reading, not compiled, not run on hardware
//
#include <hip/hip_runtime.h>
#include <math.h>

#define N_NODES 20000
#define N_EDGES 320000
#define DIM     128
#define MU_MAXF 1.0f
#define PD_ITERS 12

typedef _Float16 half_t;
typedef __attribute__((ext_vector_type(16))) _Float16 v16h;
typedef __attribute__((ext_vector_type(8)))  float    v8f;

// ---------------- math helpers ----------------
__device__ __forceinline__ float softplusf(float x) {
  return (x > 20.f) ? x : log1pf(__expf(x));
}
__device__ __forceinline__ float sigmoidf(float x) {
  return 1.f / (1.f + __expf(-x));
}
__device__ __forceinline__ float geluf(float x) {
  return 0.5f * x * (1.f + erff(x * 0.7071067811865475f));
}

// ---------------- WMMA fragment helpers (CDNA5 wave32 layouts) ----------------
// A (16x32 f16): lane<16 holds row=lane, K in {kb..kb+7, kb+16..kb+23} with kb=0;
// lane>=16 same rows with kb=8.  Two contiguous 8-half runs per lane.
__device__ __forceinline__ v16h a_frag_lds(const half_t* base, int stride, int lane) {
  int row = lane & 15;
  int kb  = (lane & 16) >> 1;          // 0 or 8
  const half_t* p = base + row * stride + kb;
  v16h a;
#pragma unroll
  for (int j = 0; j < 8; ++j) { a[j] = p[j]; a[j + 8] = p[j + 16]; }
  return a;
}
// B fragments pre-swizzled in global: frag (ks, nt, lane) = 16 contiguous halfs.
__device__ __forceinline__ v16h b_frag(const half_t* bsw, int ks, int nt, int numNT, int lane) {
  return ((const v16h*)bsw)[(size_t)(ks * numNT + nt) * 32 + lane];
}
__device__ __forceinline__ v8f wmma16(v16h a, v16h b, v8f c) {
  return __builtin_amdgcn_wmma_f32_16x16x32_f16(false, a, false, b, (short)0, c, false, false);
}

// ---------------- mean aggregation ----------------
__global__ __launch_bounds__(256)
void scatter_add_kernel(const float* __restrict__ xin, const int* __restrict__ ei,
                        float* __restrict__ acc) {
  int idx = blockIdx.x * blockDim.x + threadIdx.x;     // E*32 threads, 4 dims each
  int e = idx >> 5;
  if (e >= N_EDGES) return;
  int d0 = (idx & 31) * 4;
  int s = ei[e], t = ei[N_EDGES + e];
  const float4 v = *(const float4*)(xin + (size_t)s * DIM + d0);
  atomicAdd(&acc[(size_t)t * DIM + d0 + 0], v.x);
  atomicAdd(&acc[(size_t)t * DIM + d0 + 1], v.y);
  atomicAdd(&acc[(size_t)t * DIM + d0 + 2], v.z);
  atomicAdd(&acc[(size_t)t * DIM + d0 + 3], v.w);
}

__global__ __launch_bounds__(256)
void divide_deg_kernel(float* __restrict__ acc, const float* __restrict__ deg) {
  int i = blockIdx.x * blockDim.x + threadIdx.x;
  if (i < N_NODES * DIM) acc[i] /= deg[i >> 7];
}

// ---------------- weight prep: f32 [R][K] row-major -> f16 B-fragment-swizzled ----------------
// B layout (32x16 f16): lane<16 holds col=lane&15, K = {0..7, 16..23}; lane>=16 K = {8..15, 24..31}.
__global__ __launch_bounds__(256)
void prep_bswz_kernel(const float* __restrict__ W, int R, int Kloc, int Npad,
                      int ks_offset, int numNT,
                      const float* __restrict__ scale_logit, int scale_idx,
                      half_t* __restrict__ out) {
  int idx = blockIdx.x * blockDim.x + threadIdx.x;
  int total = Kloc * Npad;
  if (idx >= total) return;
  int j    = idx & 15;
  int lane = (idx >> 4) & 31;
  int rest = idx >> 9;
  int nt     = rest % numNT;
  int ks_loc = rest / numNT;
  int k = ks_loc * 32 + ((lane >> 4) << 3) + (j & 7) + ((j >> 3) << 4);   // local K
  int n = nt * 16 + (lane & 15);
  float val = (n < R) ? W[(size_t)n * Kloc + k] : 0.f;
  if (scale_logit) val *= 2.f * sigmoidf(scale_logit[scale_idx]);
  int ks_g = ks_offset + ks_loc;
  out[((size_t)((ks_g * numNT + nt) * 32 + lane) << 4) + j] = (half_t)val;
}

// ---------------- node GEMM ([16,512]x[512,128]) + LayerNorm ----------------
__global__ __launch_bounds__(128)
void node_gemm_ln_kernel(const float* __restrict__ x, const float* __restrict__ nb1,
                         const float* __restrict__ nb2, const half_t* __restrict__ wcat_sw,
                         const float* __restrict__ lin_bias, const float* __restrict__ ln_gamma,
                         const float* __restrict__ ln_beta, float* __restrict__ z_out) {
  __shared__ half_t feat[16 * 512];
  __shared__ float  zl[16 * 128];
  __shared__ float  mrow[16], vrow[16];
  int tile = blockIdx.x;
  int t = threadIdx.x;
  for (int idx = t; idx < 16 * 512; idx += 128) {
    int row = idx >> 9, k = idx & 511;
    int node = tile * 16 + row;
    int blk = k >> 7, kk = k & 127;
    float v;
    if (blk == 0)      v = x[(size_t)node * DIM + kk];
    else if (blk == 1) v = nb1[(size_t)node * DIM + kk];
    else if (blk == 2) v = x[(size_t)node * DIM + kk] - nb1[(size_t)node * DIM + kk];
    else               v = nb2[(size_t)node * DIM + kk];
    feat[idx] = (half_t)v;
  }
  __syncthreads();
  int wave = t >> 5, lane = t & 31;
  for (int w2 = 0; w2 < 2; ++w2) {
    int nt = wave + w2 * 4;
    v8f acc = {};
    for (int ks = 0; ks < 16; ++ks) {
      v16h a = a_frag_lds(feat + ks * 32, 512, lane);
      v16h b = b_frag(wcat_sw, ks, nt, 8, lane);
      acc = wmma16(a, b, acc);
    }
    int coln = nt * 16 + (lane & 15);
    int rbase = (lane < 16) ? 0 : 8;
#pragma unroll
    for (int i = 0; i < 8; ++i)
      zl[(rbase + i) * 128 + coln] = acc[i] + lin_bias[coln];
  }
  __syncthreads();
  if (t < 16) {
    float m = 0.f;
    for (int k = 0; k < 128; ++k) m += zl[t * 128 + k];
    m *= (1.f / 128.f);
    float v = 0.f;
    for (int k = 0; k < 128; ++k) { float d = zl[t * 128 + k] - m; v += d * d; }
    v *= (1.f / 128.f);
    mrow[t] = m; vrow[t] = v;
  }
  __syncthreads();
  for (int idx = t; idx < 2048; idx += 128) {
    int row = idx >> 7, col = idx & 127;
    float zn = (zl[idx] - mrow[row]) * rsqrtf(vrow[row] + 1e-5f) * ln_gamma[col] + ln_beta[col];
    z_out[(size_t)(tile * 16 + row) * DIM + col] = zn;
  }
}

// ---------------- edge MLPs + diff-proj + mu / raw_w ----------------
__global__ __launch_bounds__(128)
void edge_mlp_kernel(const float* __restrict__ z, const int* __restrict__ ei,
                     const half_t* __restrict__ ew1, const half_t* __restrict__ ew2,
                     const half_t* __restrict__ ew3, const half_t* __restrict__ on1,
                     const half_t* __restrict__ on2, const half_t* __restrict__ on3,
                     const half_t* __restrict__ dps,
                     const float* __restrict__ ew_b1, const float* __restrict__ ew_b2,
                     const float* __restrict__ ew_b3, const float* __restrict__ on_b1,
                     const float* __restrict__ on_b2, const float* __restrict__ on_b3,
                     const float* __restrict__ relations, const float* __restrict__ on_raw_scale,
                     float* __restrict__ raw_w, float* __restrict__ raw_w_sum,
                     float* __restrict__ mu_out) {
  __shared__ half_t fij[16 * 384];      // [s | d | p]
  __shared__ half_t fjimid[16 * 128];   // -d
  __shared__ half_t fpairmid[16 * 128]; // |d|
  __shared__ half_t h1[3 * 16 * 128];   // ew, ij, ji (post-gelu layer1)
  __shared__ half_t h2[3 * 16 * 128];   // post-gelu layer2
  __shared__ float  out3[3 * 16 * 16];
  __shared__ float  mubuf[16 * 128];
  __shared__ float  delta[16 * 8];
  __shared__ int    ridx[16], cidx[16];
  __shared__ float  wsum_part;

  int t = threadIdx.x, wave = t >> 5, lane = t & 31;
  int tile = blockIdx.x;
  if (t < 16) { ridx[t] = ei[tile * 16 + t]; cidx[t] = ei[N_EDGES + tile * 16 + t]; }
  if (t == 0) wsum_part = 0.f;
  __syncthreads();

  // feature build
  for (int idx = t; idx < 2048; idx += 128) {
    int m = idx >> 7, k = idx & 127;
    float a = z[(size_t)ridx[m] * DIM + k];
    float b = z[(size_t)cidx[m] * DIM + k];
    float s = a + b, d = a - b, p = a * b;
    fij[m * 384 + k]       = (half_t)s;
    fij[m * 384 + 128 + k] = (half_t)d;
    fij[m * 384 + 256 + k] = (half_t)p;
    fpairmid[m * 128 + k]  = (half_t)fabsf(d);
    fjimid[m * 128 + k]    = (half_t)(-d);
  }
  __syncthreads();

  // layer 1: 3 paths x 8 N-tiles, K = 384
  for (int job = wave; job < 24; job += 4) {
    int path = job >> 3, nt = job & 7;
    const half_t* B    = (path == 0) ? ew1 : on1;
    const float*  bias = (path == 0) ? ew_b1 : on_b1;
    v8f acc = {};
    for (int ks = 0; ks < 12; ++ks) {
      const half_t* Abase; int stride;
      if (ks >= 4 && ks < 8 && path != 1) {
        Abase = ((path == 0) ? fpairmid : fjimid) + (ks - 4) * 32; stride = 128;
      } else {
        Abase = fij + ks * 32; stride = 384;
      }
      v16h a = a_frag_lds(Abase, stride, lane);
      v16h b = b_frag(B, ks, nt, 8, lane);
      acc = wmma16(a, b, acc);
    }
    int coln = nt * 16 + (lane & 15);
    int rbase = (lane < 16) ? 0 : 8;
    half_t* H = h1 + path * 2048;
#pragma unroll
    for (int i = 0; i < 8; ++i)
      H[(rbase + i) * 128 + coln] = (half_t)geluf(acc[i] + bias[coln]);
  }
  __syncthreads();

  // layer 2: 3 paths x 8 N-tiles, K = 128
  for (int job = wave; job < 24; job += 4) {
    int path = job >> 3, nt = job & 7;
    const half_t* A    = h1 + path * 2048;
    const half_t* B    = (path == 0) ? ew2 : on2;
    const float*  bias = (path == 0) ? ew_b2 : on_b2;
    v8f acc = {};
    for (int ks = 0; ks < 4; ++ks) {
      v16h a = a_frag_lds(A + ks * 32, 128, lane);
      v16h b = b_frag(B, ks, nt, 8, lane);
      acc = wmma16(a, b, acc);
    }
    int coln = nt * 16 + (lane & 15);
    int rbase = (lane < 16) ? 0 : 8;
    half_t* H = h2 + path * 2048;
#pragma unroll
    for (int i = 0; i < 8; ++i)
      H[(rbase + i) * 128 + coln] = (half_t)geluf(acc[i] + bias[coln]);
  }
  __syncthreads();

  // layer 3 (padded N=16) and diff-proj GEMM
  for (int job = wave; job < 11; job += 4) {
    if (job < 3) {
      int path = job;
      const half_t* A = h2 + path * 2048;
      const half_t* B = (path == 0) ? ew3 : on3;
      v8f acc = {};
      for (int ks = 0; ks < 4; ++ks) {
        v16h a = a_frag_lds(A + ks * 32, 128, lane);
        v16h b = b_frag(B, ks, 0, 1, lane);
        acc = wmma16(a, b, acc);
      }
      int coln = lane & 15;
      int rbase = (lane < 16) ? 0 : 8;
#pragma unroll
      for (int i = 0; i < 8; ++i)
        out3[path * 256 + (rbase + i) * 16 + coln] = acc[i];
    } else {
      int nt = job - 3;   // (h_r - h_c) @ diff_proj^T : A = middle block of fij
      v8f acc = {};
      for (int ks = 0; ks < 4; ++ks) {
        v16h a = a_frag_lds(fij + 128 + ks * 32, 384, lane);
        v16h b = b_frag(dps, ks, nt, 8, lane);
        acc = wmma16(a, b, acc);
      }
      int coln = nt * 16 + (lane & 15);
      int rbase = (lane < 16) ? 0 : 8;
#pragma unroll
      for (int i = 0; i < 8; ++i)
        mubuf[(rbase + i) * 128 + coln] = acc[i];
    }
  }
  __syncthreads();

  // softmax (on paths) and raw_w (ew path)
  if (t < 32) {
    int path = 1 + (t >> 4);
    int m = t & 15;
    float* o = out3 + path * 256 + m * 16;
    float v[8]; float mx = -1e30f;
    for (int r = 0; r < 8; ++r) { v[r] = o[r] + on_b3[r]; mx = fmaxf(mx, v[r]); }
    float sum = 0.f;
    for (int r = 0; r < 8; ++r) { v[r] = __expf(v[r] - mx); sum += v[r]; }
    float inv = 1.f / sum;
    for (int r = 0; r < 8; ++r) o[r] = v[r] * inv;
  } else if (t < 48) {
    int m = t - 32;
    float sc = out3[m * 16] + ew_b3[0];
    float rw = softplusf(sc) + 1e-8f;
    raw_w[tile * 16 + m] = rw;
    atomicAdd(&wsum_part, rw);
  }
  __syncthreads();
  if (t < 16) {
    for (int r = 0; r < 8; ++r)
      delta[t * 8 + r] = 0.5f * (out3[256 + t * 16 + r] - out3[512 + t * 16 + r]);
  }
  if (t == 0) atomicAdd(raw_w_sum, wsum_part);
  __syncthreads();

  float sig_on = sigmoidf(on_raw_scale[0]);
  for (int idx = t; idx < 2048; idx += 128) {
    int m = idx >> 7, n = idx & 127;
    float v0 = mubuf[idx];
#pragma unroll
    for (int r = 0; r < 8; ++r) v0 += delta[m * 8 + r] * relations[r * DIM + n];
    v0 *= sig_on;
    v0 = MU_MAXF * tanhf(v0 / MU_MAXF);
    mu_out[(size_t)(tile * 16 + m) * DIM + n] = v0;
  }
}

// ---------------- scalar parameter precompute ----------------
// P: [0..7]=a, [8..15]=beta, [16..23]=c, 24=b0, 25=psi0, 26=lam_scale, 27=sigma,
//    28=tau, 29=z_coeff, 30=g_den
__global__ void compute_params_kernel(const int* __restrict__ max_degree,
                                      const float* __restrict__ raw_a,
                                      const float* __restrict__ raw_beta,
                                      const float* __restrict__ raw_b0,
                                      const float* __restrict__ cvec,
                                      const float* __restrict__ ew_raw_scale,
                                      const float* __restrict__ raw_w_sum,
                                      float* __restrict__ P) {
  if (threadIdx.x != 0 || blockIdx.x != 0) return;
  int md = max_degree[0];
  float step = (md > 0) ? 0.9f / sqrtf(2.f * (float)md) : 0.9f;
  float sigma = step, tau = step;
  float zc = tau;           // ALPHA = 1
  float psi0 = 0.f;
  for (int k = 0; k < 8; ++k) {
    float a  = softplusf(raw_a[k]);
    float be = softplusf(raw_beta[k]) + 1e-4f;
    P[k] = a; P[8 + k] = be; P[16 + k] = cvec[k];
    psi0 += a * sigmoidf(cvec[k]);
  }
  float b0 = softplusf(raw_b0[0]);
  psi0 += b0;
  P[24] = b0; P[25] = psi0;
  float mean = raw_w_sum[0] / (float)N_EDGES;
  P[26] = softplusf(ew_raw_scale[0]) * (1.f / fmaxf(mean, 1e-6f)) / sigma;
  P[27] = sigma; P[28] = tau; P[29] = zc; P[30] = 1.f + zc;
}

// ---------------- PDHG edge step: y update + Newton prox + K^T scatter ----------------
__global__ __launch_bounds__(256)
void pdhg_edge_kernel(const float* __restrict__ u_bar, const float* __restrict__ mu,
                      const int* __restrict__ ei, const float* __restrict__ raw_w,
                      const float* __restrict__ P, float* __restrict__ y,
                      float* __restrict__ kt) {
  int wave = threadIdx.x >> 5, lane = threadIdx.x & 31;
  int e = blockIdx.x * 8 + wave;
  if (e >= N_EDGES) return;
  float sigma = P[27];
  float lam = raw_w[e] * P[26];
  int rn = ei[e], cn = ei[N_EDGES + e];
  int d0 = lane * 4;
  const float4 ur = *(const float4*)(u_bar + (size_t)rn * DIM + d0);
  const float4 uc = *(const float4*)(u_bar + (size_t)cn * DIM + d0);
  const float4 mm = *(const float4*)(mu + (size_t)e * DIM + d0);
  const float4 yy = *(const float4*)(y + (size_t)e * DIM + d0);
  float yt[4];
  yt[0] = yy.x + sigma * (ur.x - uc.x - mm.x);
  yt[1] = yy.y + sigma * (ur.y - uc.y - mm.y);
  yt[2] = yy.z + sigma * (ur.z - uc.z - mm.z);
  yt[3] = yy.w + sigma * (ur.w - uc.w - mm.w);
  float inv_sig = 1.f / sigma;
  float r2 = 0.f;
#pragma unroll
  for (int j = 0; j < 4; ++j) { float v = yt[j] * inv_sig; r2 += v * v; }
#pragma unroll
  for (int off = 16; off; off >>= 1) r2 += __shfl_xor(r2, off, 32);
  float r = sqrtf(r2);
  float b0 = P[24], psi0 = P[25];
  bool active = (r > 1e-8f) && (lam * psi0 - r < 0.f);
  float s = active ? r : 0.f;
#pragma unroll
  for (int it = 0; it < 8; ++it) {
    float ps = b0, pp = 0.f;
#pragma unroll
    for (int k = 0; k < 8; ++k) {
      float a = P[k], be = P[8 + k], cc = P[16 + k];
      float sg = sigmoidf(s * be + cc);
      ps += a * sg;
      pp += a * be * sg * (1.f - sg);
    }
    float sn = s - (s + lam * ps - r) / (1.f + lam * pp);
    sn = fminf(fmaxf(sn, 0.f), r);
    s = active ? sn : s;
  }
  float fac = (r > 1e-8f) ? (s / r) : 0.f;
  float om = 1.f - fac;          // y_next = (1 - scale) * y_tilde (Moreau)
  float4 yn;
  yn.x = om * yt[0]; yn.y = om * yt[1]; yn.z = om * yt[2]; yn.w = om * yt[3];
  *(float4*)(y + (size_t)e * DIM + d0) = yn;
  atomicAdd(&kt[(size_t)rn * DIM + d0 + 0], yn.x);
  atomicAdd(&kt[(size_t)rn * DIM + d0 + 1], yn.y);
  atomicAdd(&kt[(size_t)rn * DIM + d0 + 2], yn.z);
  atomicAdd(&kt[(size_t)rn * DIM + d0 + 3], yn.w);
  atomicAdd(&kt[(size_t)cn * DIM + d0 + 0], -yn.x);
  atomicAdd(&kt[(size_t)cn * DIM + d0 + 1], -yn.y);
  atomicAdd(&kt[(size_t)cn * DIM + d0 + 2], -yn.z);
  atomicAdd(&kt[(size_t)cn * DIM + d0 + 3], -yn.w);
}

// ---------------- PDHG node step ----------------
__global__ __launch_bounds__(256)
void pdhg_node_kernel(const float* __restrict__ z, const float* __restrict__ P,
                      float* __restrict__ u, float* __restrict__ u_bar,
                      float* __restrict__ kt) {
  int i = blockIdx.x * blockDim.x + threadIdx.x;
  if (i >= N_NODES * DIM) return;
  float tau = P[28], zc = P[29], inv_gd = 1.f / P[30];
  float uo = u[i];
  float un = (uo - tau * kt[i] + zc * z[i]) * inv_gd;
  u[i] = un;
  u_bar[i] = un + (un - uo);   // XI = 1
  kt[i] = 0.f;                 // reset for next iteration
}

// ---------------- host side ----------------
extern "C" void kernel_launch(void* const* d_in, const int* in_sizes, int n_in,
                              void* d_out, int out_size, void* d_ws, size_t ws_size,
                              hipStream_t stream) {
  (void)in_sizes; (void)n_in; (void)out_size; (void)ws_size;
  const float* x            = (const float*)d_in[0];
  const int*   ei_mp        = (const int*)d_in[1];
  const float* deg_mp       = (const float*)d_in[2];
  const int*   ei_sv        = (const int*)d_in[3];
  const int*   max_deg      = (const int*)d_in[4];
  const float* W_self       = (const float*)d_in[5];
  const float* W_nb1        = (const float*)d_in[6];
  const float* W_hp         = (const float*)d_in[7];
  const float* W_nb2        = (const float*)d_in[8];
  const float* lin_bias     = (const float*)d_in[9];
  const float* branch_logit = (const float*)d_in[10];
  const float* ln_gamma     = (const float*)d_in[11];
  const float* ln_beta      = (const float*)d_in[12];
  const float* raw_b0       = (const float*)d_in[13];
  const float* raw_a        = (const float*)d_in[14];
  const float* raw_beta     = (const float*)d_in[15];
  const float* cvec         = (const float*)d_in[16];
  const float* ew_w1        = (const float*)d_in[17];
  const float* ew_b1        = (const float*)d_in[18];
  const float* ew_w2        = (const float*)d_in[19];
  const float* ew_b2        = (const float*)d_in[20];
  const float* ew_w3        = (const float*)d_in[21];
  const float* ew_b3        = (const float*)d_in[22];
  const float* ew_raw_scale = (const float*)d_in[23];
  const float* relations    = (const float*)d_in[24];
  const float* on_w1        = (const float*)d_in[25];
  const float* on_b1        = (const float*)d_in[26];
  const float* on_w2        = (const float*)d_in[27];
  const float* on_b2        = (const float*)d_in[28];
  const float* on_w3        = (const float*)d_in[29];
  const float* on_b3        = (const float*)d_in[30];
  const float* diff_proj    = (const float*)d_in[31];
  const float* on_raw_scale = (const float*)d_in[32];

  const size_t NODE_BYTES = (size_t)N_NODES * DIM * sizeof(float);
  const size_t EDGE_BYTES = (size_t)N_EDGES * DIM * sizeof(float);

  char* base = (char*)d_ws;
  size_t off = 0;
  auto carve = [&](size_t bytes) -> void* {
    void* p = base + off;
    off = (off + bytes + 255) & ~(size_t)255;
    return p;
  };
  float*  nb1    = (float*)carve(NODE_BYTES);
  float*  nb2    = (float*)carve(NODE_BYTES);
  float*  z      = (float*)carve(NODE_BYTES);
  float*  u      = (float*)carve(NODE_BYTES);
  float*  u_bar  = (float*)carve(NODE_BYTES);
  float*  kt     = (float*)carve(NODE_BYTES);
  half_t* wcat   = (half_t*)carve((size_t)512 * 128 * 2);
  half_t* ew1    = (half_t*)carve((size_t)384 * 128 * 2);
  half_t* ew2    = (half_t*)carve((size_t)128 * 128 * 2);
  half_t* ew3    = (half_t*)carve((size_t)128 * 16 * 2);
  half_t* on1    = (half_t*)carve((size_t)384 * 128 * 2);
  half_t* on2    = (half_t*)carve((size_t)128 * 128 * 2);
  half_t* on3    = (half_t*)carve((size_t)128 * 16 * 2);
  half_t* dps    = (half_t*)carve((size_t)128 * 128 * 2);
  float*  raw_w  = (float*)carve((size_t)N_EDGES * sizeof(float));
  float*  rsum   = (float*)carve(256);
  float*  params = (float*)carve(256);
  float*  mu     = (float*)carve(EDGE_BYTES);
  float*  y      = (float*)carve(EDGE_BYTES);

  // --- mean aggregation (2 hops) ---
  hipMemsetAsync(nb1, 0, NODE_BYTES, stream);
  scatter_add_kernel<<<(N_EDGES * 32 + 255) / 256, 256, 0, stream>>>(x, ei_mp, nb1);
  divide_deg_kernel<<<(N_NODES * DIM + 255) / 256, 256, 0, stream>>>(nb1, deg_mp);
  hipMemsetAsync(nb2, 0, NODE_BYTES, stream);
  scatter_add_kernel<<<(N_EDGES * 32 + 255) / 256, 256, 0, stream>>>(nb1, ei_mp, nb2);
  divide_deg_kernel<<<(N_NODES * DIM + 255) / 256, 256, 0, stream>>>(nb2, deg_mp);

  // --- f16 weight prep (transpose + fragment swizzle) ---
  auto prep = [&](const float* W, int R, int Kloc, int Npad, int ks_off, int numNT,
                  const float* sl, int si, half_t* outp) {
    int total = Kloc * Npad;
    prep_bswz_kernel<<<(total + 255) / 256, 256, 0, stream>>>(W, R, Kloc, Npad, ks_off,
                                                              numNT, sl, si, outp);
  };
  prep(W_self, 128, 128, 128, 0, 8, branch_logit, 0, wcat);
  prep(W_nb1, 128, 128, 128, 4, 8, branch_logit, 1, wcat);
  prep(W_hp, 128, 128, 128, 8, 8, branch_logit, 2, wcat);
  prep(W_nb2, 128, 128, 128, 12, 8, branch_logit, 3, wcat);
  prep(ew_w1, 128, 384, 128, 0, 8, nullptr, 0, ew1);
  prep(ew_w2, 128, 128, 128, 0, 8, nullptr, 0, ew2);
  prep(ew_w3, 1, 128, 16, 0, 1, nullptr, 0, ew3);
  prep(on_w1, 128, 384, 128, 0, 8, nullptr, 0, on1);
  prep(on_w2, 128, 128, 128, 0, 8, nullptr, 0, on2);
  prep(on_w3, 8, 128, 16, 0, 1, nullptr, 0, on3);
  prep(diff_proj, 128, 128, 128, 0, 8, nullptr, 0, dps);

  // --- node GEMM + LayerNorm -> z ---
  node_gemm_ln_kernel<<<N_NODES / 16, 128, 0, stream>>>(x, nb1, nb2, wcat, lin_bias,
                                                        ln_gamma, ln_beta, z);

  // --- edge MLPs -> raw_w, mu ---
  hipMemsetAsync(rsum, 0, sizeof(float), stream);
  edge_mlp_kernel<<<N_EDGES / 16, 128, 0, stream>>>(
      z, ei_sv, ew1, ew2, ew3, on1, on2, on3, dps, ew_b1, ew_b2, ew_b3, on_b1, on_b2,
      on_b3, relations, on_raw_scale, raw_w, rsum, mu);

  // --- scalar params ---
  compute_params_kernel<<<1, 1, 0, stream>>>(max_deg, raw_a, raw_beta, raw_b0, cvec,
                                             ew_raw_scale, rsum, params);

  // --- PDHG init: u = u_bar = z, y = 0, kt = 0 ---
  hipMemcpyAsync(u, z, NODE_BYTES, hipMemcpyDeviceToDevice, stream);
  hipMemcpyAsync(u_bar, z, NODE_BYTES, hipMemcpyDeviceToDevice, stream);
  hipMemsetAsync(y, 0, EDGE_BYTES, stream);
  hipMemsetAsync(kt, 0, NODE_BYTES, stream);

  for (int it = 0; it < PD_ITERS; ++it) {
    pdhg_edge_kernel<<<N_EDGES / 8, 256, 0, stream>>>(u_bar, mu, ei_sv, raw_w, params, y, kt);
    pdhg_node_kernel<<<(N_NODES * DIM + 255) / 256, 256, 0, stream>>>(z, params, u, u_bar, kt);
  }
  hipMemcpyAsync(d_out, u, NODE_BYTES, hipMemcpyDeviceToDevice, stream);
}
